// Mamba_81217831567946
// MI455X (gfx1250) — matrix operand order repaired and checked by reading, not compiled, not run on hardware
//
#include <hip/hip_runtime.h>
#include <cstdint>
#include <cstddef>

// Model dims (match reference)
#define MDIM   768
#define VDIM   32000
#define RDIM   48
#define NSTATE 16
#define KCONV  4
#define LAYERS 2
#define HDIM   1536
#define BD     2
#define SD     1024
#define BSROWS (BD * SD)          // 2048
#define TWOH   (2 * HDIM)         // 3072
#define XDBLW  (RDIM + 2*NSTATE)  // 80

typedef __attribute__((ext_vector_type(16))) _Float16 v16h;
typedef __attribute__((ext_vector_type(8)))  float    v8f;
typedef __attribute__((ext_vector_type(4)))  _Float16 half4;

#define TILE_M 64
#define TILE_N 128
#define TILE_K 64
#define LDS_PAD 8

// ---------------------------------------------------------------------------
// Generic GEMM: C[Mr x Nc] = A[Mr x K] (row stride lda) * W[Nc x K]^T + bias
// act: 0 = none, 1 = softplus.  f16 inputs via LDS, f32 WMMA accumulation.
// Block = 256 threads (8 waves); block tile 64x128; wave tile 32x32.
// FAST: shapes aligned (Mr%64==0, Nc%128==0, K%64==0, lda==K) -> float4
// staging, no guards.  Slow path: clamped-address loads + select (no
// divergent branches, loads pipeline).
// ---------------------------------------------------------------------------
template <bool FAST>
__global__ __launch_bounds__(256) void gemm_wmma_kernel(
    const float* __restrict__ A, int lda,
    const float* __restrict__ W,          // Nc x K row-major
    const float* __restrict__ bias,       // Nc (may be null)
    float* __restrict__ C,                // Mr x Nc
    int Mr, int Nc, int K, int act)
{
  __shared__ _Float16 As[TILE_M][TILE_K + LDS_PAD];
  __shared__ _Float16 Ws[TILE_N][TILE_K + LDS_PAD];

  const int bm   = blockIdx.x * TILE_M;
  const int bn   = blockIdx.y * TILE_N;
  const int tid  = threadIdx.x;
  const int lane = tid & 31;
  const int wave = tid >> 5;           // 0..7
  const int wrow = (wave >> 2) * 32;   // 0 or 32  (row band)
  const int wcol = (wave & 3) * 32;    // 0,32,64,96 (col band)
  const int half = lane >> 4;          // lane half (0/1)
  const int l16  = lane & 15;

  v8f acc[2][2] = {{{}, {}}, {{}, {}}};

  for (int k0 = 0; k0 < K; k0 += TILE_K) {
    if (FAST && (k0 + TILE_K < K)) {
      // Speculative prefetch of next K panel (global_prefetch_b8)
      __builtin_prefetch(&A[(size_t)(bm + (tid >> 2)) * lda + k0 + TILE_K], 0, 1);
      __builtin_prefetch(&W[(size_t)(bn + (tid >> 1)) * K   + k0 + TILE_K], 0, 1);
    }
    if (FAST) {
      // A tile: 64x64 floats = 1024 float4, 4 per thread
      #pragma unroll
      for (int j = 0; j < 4; ++j) {
        int idx = tid + j * 256;
        int r = idx >> 4, c4 = (idx & 15) << 2;
        float4 v = *(const float4*)&A[(size_t)(bm + r) * lda + (k0 + c4)];
        half4 h = {(_Float16)v.x, (_Float16)v.y, (_Float16)v.z, (_Float16)v.w};
        *(half4*)&As[r][c4] = h;
      }
      // W tile: 128x64 floats = 2048 float4, 8 per thread
      #pragma unroll
      for (int j = 0; j < 8; ++j) {
        int idx = tid + j * 256;
        int r = idx >> 4, c4 = (idx & 15) << 2;
        float4 v = *(const float4*)&W[(size_t)(bn + r) * K + (k0 + c4)];
        half4 h = {(_Float16)v.x, (_Float16)v.y, (_Float16)v.z, (_Float16)v.w};
        *(half4*)&Ws[r][c4] = h;
      }
    } else {
      // Clamped-address loads + select: unconditional, pipelined.
      #pragma unroll
      for (int j = 0; j < 16; ++j) {           // 64*64/256
        int i = tid + j * 256;
        int r = i >> 6, c = i & 63;
        int gr = bm + r, gc = k0 + c;
        bool ok = (gr < Mr) && (gc < K);
        int grc = min(gr, Mr - 1), gcc = min(gc, K - 1);
        float v = A[(size_t)grc * lda + gcc];
        As[r][c] = (_Float16)(ok ? v : 0.f);
      }
      #pragma unroll
      for (int j = 0; j < 32; ++j) {           // 128*64/256
        int i = tid + j * 256;
        int r = i >> 6, c = i & 63;
        int gr = bn + r, gc = k0 + c;
        bool ok = (gr < Nc) && (gc < K);
        int grc = min(gr, Nc - 1), gcc = min(gc, K - 1);
        float v = W[(size_t)grc * K + gcc];
        Ws[r][c] = (_Float16)(ok ? v : 0.f);
      }
    }
    __syncthreads();

    #pragma unroll
    for (int kk = 0; kk < TILE_K; kk += 32) {
      // A fragments (contiguous k order, same as B):
      // lanes 0-15 -> K kk+0..15; lanes 16-31 -> K kk+16..31.
      v16h af[2];
      #pragma unroll
      for (int mt = 0; mt < 2; ++mt) {
        const int am = wrow + mt * 16 + l16;
        #pragma unroll
        for (int e = 0; e < 16; ++e) {
          int k = kk + half * 16 + e;
          af[mt][e] = As[am][k];
        }
      }
      // B fragments: lane holds column, lanes 0-15 K kk..kk+15, 16-31 +16.
      v16h bf[2];
      #pragma unroll
      for (int nt = 0; nt < 2; ++nt) {
        const int cn = wcol + nt * 16 + l16;
        #pragma unroll
        for (int e = 0; e < 16; ++e) {
          bf[nt][e] = Ws[cn][kk + half * 16 + e];
        }
      }
      #pragma unroll
      for (int mt = 0; mt < 2; ++mt)
        #pragma unroll
        for (int nt = 0; nt < 2; ++nt)
          acc[mt][nt] = __builtin_amdgcn_wmma_f32_16x16x32_f16(
              false, af[mt], false, bf[nt], (short)0, acc[mt][nt], false, false);
    }
    __syncthreads();
  }

  // D layout: lane -> N = l16, M = half*8 + reg
  #pragma unroll
  for (int mt = 0; mt < 2; ++mt) {
    #pragma unroll
    for (int nt = 0; nt < 2; ++nt) {
      #pragma unroll
      for (int r = 0; r < 8; ++r) {
        int gm = bm + wrow + mt * 16 + half * 8 + r;
        int gn = bn + wcol + nt * 16 + l16;
        if (FAST || (gm < Mr && gn < Nc)) {
          float v = acc[mt][nt][r] + (bias ? bias[gn] : 0.f);
          if (act == 1) v = (v > 20.f) ? v : logf(1.f + expf(v));
          C[(size_t)gm * Nc + gn] = v;
        }
      }
    }
  }
}

// ---------------------------------------------------------------------------
// Embedding gather (float4): x[bs, :] = emb[ids[bs], :]
// ---------------------------------------------------------------------------
__global__ void embed_kernel(const int* __restrict__ ids,
                             const float* __restrict__ emb,
                             float* __restrict__ x, int total4)
{
  int i = blockIdx.x * blockDim.x + threadIdx.x;
  if (i < total4) {
    const int m4 = MDIM / 4;
    int row = i / m4, c4 = i - row * m4;
    ((float4*)x)[i] = ((const float4*)(emb + (size_t)ids[row] * MDIM))[c4];
  }
}

// ---------------------------------------------------------------------------
// Depthwise causal conv (K=4) over the first H columns of xr, + bias, SiLU
// ---------------------------------------------------------------------------
__global__ void conv_silu_kernel(const float* __restrict__ xr,
                                 const float* __restrict__ cw,  // H*KCONV
                                 const float* __restrict__ cb,  // H
                                 float* __restrict__ xh, int total)
{
  int i = blockIdx.x * blockDim.x + threadIdx.x;
  if (i >= total) return;
  int h  = i % HDIM;
  int bs = i / HDIM;
  int s  = bs % SD;
  int b  = bs / SD;
  float acc = cb[h];
  #pragma unroll
  for (int k = 0; k < KCONV; ++k) {
    int sp = s - (KCONV - 1) + k;
    if (sp >= 0)
      acc += cw[h * KCONV + k] * xr[(size_t)(b * SD + sp) * TWOH + h];
  }
  xh[i] = acc / (1.f + expf(-acc));  // SiLU
}

// ---------------------------------------------------------------------------
// Selective scan: one (b,h) chain per 16-lane group; lane = state index n.
// x_dbl columns: [48:64)=Bm, [64:80)=Cm.  y = sum_n(st_n * Cm_n) + u*D.
// Grid sized exactly: BD*HDIM*16 threads.
// ---------------------------------------------------------------------------
__global__ __launch_bounds__(256) void scan_kernel(
    const float* __restrict__ u,     // (B*S, H)
    const float* __restrict__ delta, // (B*S, H)
    const float* __restrict__ xdbl,  // (B*S, 80)
    const float* __restrict__ Dp,    // (H)
    float* __restrict__ y)           // (B*S, H)
{
  int gid   = blockIdx.x * blockDim.x + threadIdx.x;
  int group = gid >> 4;            // chain id over B*H
  int n     = gid & 15;            // state index
  int b = group / HDIM, h = group % HDIM;
  const float An = -(float)(n + 1) - 1e-15f;
  const float Dh = Dp[h];
  float st = 0.f;
  for (int s = 0; s < SD; ++s) {
    size_t row = (size_t)(b * SD + s);
    float dt = delta[row * HDIM + h];
    float uu = u[row * HDIM + h];
    float bm = xdbl[row * XDBLW + RDIM + n];
    float cm = xdbl[row * XDBLW + RDIM + NSTATE + n];
    st = expf(dt * An) * st + dt * uu * bm;
    float pv = st * cm;
    pv += __shfl_xor(pv, 1, 16);
    pv += __shfl_xor(pv, 2, 16);
    pv += __shfl_xor(pv, 4, 16);
    pv += __shfl_xor(pv, 8, 16);
    if (n == 0) y[row * HDIM + h] = pv + uu * Dh;
  }
}

// ---------------------------------------------------------------------------
// y *= silu(residual), residual = xr[:, H + h]
// ---------------------------------------------------------------------------
__global__ void gate_kernel(float* __restrict__ y,
                            const float* __restrict__ xr, int total)
{
  int i = blockIdx.x * blockDim.x + threadIdx.x;
  if (i >= total) return;
  int h = i % HDIM, bs = i / HDIM;
  float r = xr[(size_t)bs * TWOH + HDIM + h];
  y[i] *= r / (1.f + expf(-r));
}

// ---------------------------------------------------------------------------
// out[row] = (resid ? resid[row] : 0) + rmsnorm(in[row]) * w ;  1 block/row
// ---------------------------------------------------------------------------
__global__ __launch_bounds__(256) void rmsnorm_kernel(
    const float* __restrict__ in, const float* __restrict__ w,
    const float* __restrict__ resid, float* __restrict__ out)
{
  __shared__ float red[8];
  const int row  = blockIdx.x;
  const int tid  = threadIdx.x;
  const int lane = tid & 31;
  const int wv   = tid >> 5;
  const float* p = in + (size_t)row * MDIM;

  float ss = 0.f;
  for (int c = tid; c < MDIM; c += 256) { float v = p[c]; ss += v * v; }
  #pragma unroll
  for (int off = 16; off > 0; off >>= 1) ss += __shfl_xor(ss, off, 32);
  if (lane == 0) red[wv] = ss;
  __syncthreads();
  if (tid == 0) {
    float t = 0.f;
    #pragma unroll
    for (int j = 0; j < 8; ++j) t += red[j];
    red[0] = t;
  }
  __syncthreads();
  const float scale = rsqrtf(red[0] / (float)MDIM + 1e-15f);
  for (int c = tid; c < MDIM; c += 256) {
    float v = p[c] * scale * w[c];
    if (resid) v += resid[(size_t)row * MDIM + c];
    out[(size_t)row * MDIM + c] = v;
  }
}

// ---------------------------------------------------------------------------
extern "C" void kernel_launch(void* const* d_in, const int* in_sizes, int n_in,
                              void* d_out, int out_size, void* d_ws, size_t ws_size,
                              hipStream_t stream)
{
  (void)in_sizes; (void)n_in; (void)out_size; (void)ws_size;

  const int*   ids   = (const int*)  d_in[0];
  const float* emb   = (const float*)d_in[1];
  const float* w_in  = (const float*)d_in[2];   // (L, 2H, M)
  const float* b_in  = (const float*)d_in[3];   // (L, 2H)
  const float* cw    = (const float*)d_in[4];   // (L, H, 1, K)
  const float* cb    = (const float*)d_in[5];   // (L, H)
  const float* w_x   = (const float*)d_in[6];   // (L, 80, H)
  const float* b_x   = (const float*)d_in[7];   // (L, 80)
  const float* w_dt  = (const float*)d_in[8];   // (L, H, R)
  const float* b_dt  = (const float*)d_in[9];   // (L, H)
  const float* w_out = (const float*)d_in[10];  // (L, M, H)
  const float* b_out = (const float*)d_in[11];  // (L, M)
  const float* Dp    = (const float*)d_in[12];  // (L, H)
  const float* bnw   = (const float*)d_in[13];  // (L, M)
  const float* fnw   = (const float*)d_in[14];  // (M)
  const float* hw    = (const float*)d_in[15];  // (V, M)
  const float* hb    = (const float*)d_in[16];  // (V)
  float* out = (float*)d_out;

  // Workspace layout (floats)
  float* ws  = (float*)d_ws;
  float* x    = ws;                           // BS*M
  float* xr   = x    + (size_t)BSROWS * MDIM; // BS*2H
  float* xh   = xr   + (size_t)BSROWS * TWOH; // BS*H
  float* xdbl = xh   + (size_t)BSROWS * HDIM; // BS*80
  float* dlt  = xdbl + (size_t)BSROWS * XDBLW;// BS*H
  float* ybuf = dlt  + (size_t)BSROWS * HDIM; // BS*H
  float* yout = ybuf + (size_t)BSROWS * HDIM; // BS*M
  float* xn   = yout + (size_t)BSROWS * MDIM; // BS*M

  auto gemm = [&](const float* Ap, int lda, const float* Wp, const float* bp,
                  float* Cp, int Mr, int Nc, int Kd, int act) {
    dim3 grid((Mr + TILE_M - 1) / TILE_M, (Nc + TILE_N - 1) / TILE_N);
    bool fast = (lda == Kd) && (Mr % TILE_M == 0) && (Nc % TILE_N == 0) &&
                (Kd % TILE_K == 0);
    if (fast)
      gemm_wmma_kernel<true><<<grid, 256, 0, stream>>>(Ap, lda, Wp, bp, Cp,
                                                       Mr, Nc, Kd, act);
    else
      gemm_wmma_kernel<false><<<grid, 256, 0, stream>>>(Ap, lda, Wp, bp, Cp,
                                                        Mr, Nc, Kd, act);
  };

  // Embedding (float4)
  {
    int total4 = BSROWS * (MDIM / 4);
    embed_kernel<<<(total4 + 255) / 256, 256, 0, stream>>>(ids, emb, x, total4);
  }

  for (int l = 0; l < LAYERS; ++l) {
    const float* wi = w_in  + (size_t)l * TWOH * MDIM;
    const float* bi = b_in  + (size_t)l * TWOH;
    const float* wc = cw    + (size_t)l * HDIM * KCONV;
    const float* bc = cb    + (size_t)l * HDIM;
    const float* wx = w_x   + (size_t)l * XDBLW * HDIM;
    const float* bx = b_x   + (size_t)l * XDBLW;
    const float* wd = w_dt  + (size_t)l * HDIM * RDIM;
    const float* bd = b_dt  + (size_t)l * HDIM;
    const float* wo = w_out + (size_t)l * MDIM * HDIM;
    const float* bo = b_out + (size_t)l * MDIM;
    const float* Dl = Dp    + (size_t)l * HDIM;
    const float* nw = bnw   + (size_t)l * MDIM;

    // xr = x @ lin_in_w^T + b           (2048 x 3072, K=768)  [FAST]
    gemm(x, MDIM, wi, bi, xr, BSROWS, TWOH, MDIM, 0);
    // depthwise causal conv + SiLU -> xh
    {
      int total = BSROWS * HDIM;
      conv_silu_kernel<<<(total + 255) / 256, 256, 0, stream>>>(xr, wc, bc, xh, total);
    }
    // x_dbl = xh @ lin_x_w^T + b        (2048 x 80, K=1536)   [slow]
    gemm(xh, HDIM, wx, bx, xdbl, BSROWS, XDBLW, HDIM, 0);
    // delta = softplus(x_dbl[:, :48] @ lin_dt_w^T + b)  (2048 x 1536, K=48) [slow]
    gemm(xdbl, XDBLW, wd, bd, dlt, BSROWS, HDIM, RDIM, 1);
    // selective scan -> ybuf (16 lanes per chain)
    {
      int total = BD * HDIM * 16;     // 49152, exact multiple of 256
      scan_kernel<<<total / 256, 256, 0, stream>>>(xh, dlt, xdbl, Dl, ybuf);
    }
    // gate with silu(residual)
    {
      int total = BSROWS * HDIM;
      gate_kernel<<<(total + 255) / 256, 256, 0, stream>>>(ybuf, xr, total);
    }
    // yout = ybuf @ lin_out_w^T + b     (2048 x 768, K=1536)  [FAST]
    gemm(ybuf, HDIM, wo, bo, yout, BSROWS, MDIM, HDIM, 0);
    // x = x + rmsnorm(yout) * block_norm_w
    rmsnorm_kernel<<<BSROWS, 256, 0, stream>>>(yout, nw, x, x);
  }

  // Final norm + head
  rmsnorm_kernel<<<BSROWS, 256, 0, stream>>>(x, fnw, nullptr, xn);
  // logits = xn @ head_w^T + head_b     (2048 x 32000, K=768) [FAST]
  gemm(xn, MDIM, hw, hb, out, BSROWS, VDIM, MDIM, 0);
}
